// PackedStdScaler_28896539967788
// MI455X (gfx1250) — compile-verified, matrix operand order checked by reading, not run on hardware
//
#include <hip/hip_runtime.h>
#include <hip/hip_bf16.h>

// PackedStdScaler for MI455X (gfx1250, wave32) — single fused kernel.
//
// Semantics: segmented mean/std over <=256 (sample,variate) groups per batch
// (reference's O(B*S^2) masked matmuls collapse to an O(B*S*D) segmented
// reduction). Total HBM traffic ~1.5MB -> ~65ns at 23.3TB/s: launch-latency
// bound, so we use ONE kernel, keep the whole batch LDS-resident, and make
// the group reduction deterministic-parallel:
//
//   Phase 0: per-position sums via v_wmma_f32_16x16x32_f16 (one WMMA per 16
//            positions yields both s1 = sum_d t*obs and cnt = sum_d obs, f32
//            accumulate), written straight to LDS (no global workspace).
//   Phase 1: sample_id is sorted -> binary-search the 8 sample ranges; group
//            (s,v) scans only its range, split 4 ways across 1024 threads,
//            partials combined in fixed order (deterministic, no float
//            atomics). Work: 32*S instead of 256*S match-tests.
//   Phase 2: q_i = sum_d (t - loc_g)^2 * obs in f32.
//   Phase 3: same range-scan for s2 -> var = safe_div(s2, cnt-1),
//            scale = sqrt(var + 1e-5).
//   Phase 4: scatter with pad override (sample_id==0 -> loc=0, scale=1).

#define B_DIM 4
#define S_DIM 4096
#define D_DIM 16
#define NGROUPS 256           // N_SAMPLES(8) * N_VARIATES(32)
#define BS_TOTAL (B_DIM * S_DIM)
#define THREADS 1024          // 32 wave32 waves
#define NPART 4               // range split per group

typedef __attribute__((ext_vector_type(16))) _Float16 v16h;
typedef __attribute__((ext_vector_type(8)))  float    v8f;

__global__ void __launch_bounds__(THREADS)
pss_fused_kernel(const float* __restrict__ target,
                 const unsigned char* __restrict__ obs,   // jax bool: 1 byte
                 const int* __restrict__ sample_id,
                 const int* __restrict__ variate_id,
                 float* __restrict__ out)                 // [2*B*S]: loc, scale
{
    __shared__ int   s_key[S_DIM];              // 16 KB  sid*32+vid
    __shared__ float s_cnt[S_DIM];              // 16 KB  per-position obs count
    __shared__ float s_val[S_DIM];              // 16 KB  per-position s1, then q
    __shared__ int   s_bound[9];                //        sample range bounds
    __shared__ float s_pcnt[NPART][NGROUPS];    //  4 KB  partial cnt
    __shared__ float s_ps[NPART][NGROUPS];      //  4 KB  partial s1 / s2
    __shared__ float g_cnt[NGROUPS];
    __shared__ float g_loc[NGROUPS];
    __shared__ float g_scale[NGROUPS];

    const int tid    = threadIdx.x;
    const int lane   = tid & 31;
    const int waveId = tid >> 5;                // 0..31
    const int boff   = blockIdx.x * S_DIM;

    // ---- stage keys (uniform trip count: EXEC stays full) ----------------
    for (int i = tid; i < S_DIM; i += THREADS)
        s_key[i] = sample_id[boff + i] * 32 + variate_id[boff + i];   // [0,256)

    // ---- Phase 0: WMMA row sums, 8 tiles of 16 positions per wave --------
    // B selector (32x16): b[k,n] = ((n<8) == (k<16)) ? 1 : 0, so
    // C[m,0:8] = sum_{k<16} A = s1_m and C[m,8:16] = sum_{k>=16} A = cnt_m.
    // 16-bit B layout mirrors A: lane -> column n=lane&15,
    // element e -> K = (e<8 ? e : 8+e) + (lane>=16 ? 8 : 0).
    const int n = lane & 15;
    v16h bm;
#pragma unroll
    for (int e = 0; e < 16; ++e) {
        const int k = ((e < 8) ? e : 8 + e) + ((lane >= 16) ? 8 : 0);
        const bool one = (n < 8) ? (k < 16) : (k >= 16);
        bm[e] = one ? (_Float16)1.0f : (_Float16)0.0f;
    }

    const int m     = lane & 15;                 // row within tile
    const int dbase = (lane & 16) ? 8 : 0;       // this lane supplies d=dbase..+7
#pragma unroll
    for (int t = 0; t < S_DIM / 16 / 32; ++t) {  // 8 tiles per wave
        const int base = (waveId * 8 + t) * 16;
        const int i    = base + m;

        const float4* tp =
            reinterpret_cast<const float4*>(target + (size_t)(boff + i) * D_DIM + dbase);
        const float4 t0 = tp[0];
        const float4 t1 = tp[1];
        const unsigned long long mk = *reinterpret_cast<const unsigned long long*>(
            obs + (size_t)(boff + i) * D_DIM + dbase);

        const float tf[8] = {t0.x, t0.y, t0.z, t0.w, t1.x, t1.y, t1.z, t1.w};

        // A (16-bit layout, ISA 7.12.2): a[e<8] = t*obs (K in [0,16)),
        // a[e>=8] = obs (K in [16,32)); both halves use d = dbase + (e&7).
        v16h a;
#pragma unroll
        for (int e = 0; e < 8; ++e) {
            const float mv = (float)((mk >> (8 * e)) & 0xffu);   // 0.0 / 1.0
            a[e]     = (_Float16)(tf[e] * mv);
            a[e + 8] = (_Float16)mv;
        }

        v8f c = {};
        c = __builtin_amdgcn_wmma_f32_16x16x32_f16(false, a, false, bm, (short)0, c,
                                                   false, false);

        // C layout: lane holds N=lane&15, VGPR r -> M = r + (lane>=16?8:0).
        // One lane per (row-half, column-kind) writes its 8 rows to LDS.
        if ((lane & 7) == 0) {
            float* dst = (lane & 8) ? s_cnt : s_val;      // col 8 -> cnt, col 0 -> s1
            const int mbase = base + ((lane & 16) ? 8 : 0);
#pragma unroll
            for (int r = 0; r < 8; ++r) dst[mbase + r] = c[r];
        }
    }
    __syncthreads();

    // ---- sample range bounds: sample_id sorted -> lower_bound per sample --
    if (tid <= 8) {
        int lo;
        if (tid == 0)      lo = 0;
        else if (tid == 8) lo = S_DIM;
        else {
            int a = 0, bnd = S_DIM;
            while (a < bnd) {                   // first i with sid[i] >= tid
                const int mid = (a + bnd) >> 1;
                if ((s_key[mid] >> 5) < tid) a = mid + 1; else bnd = mid;
            }
            lo = a;
        }
        s_bound[tid] = lo;
    }
    __syncthreads();

    // ---- Phase 1: cnt, s1 -> loc (deterministic 4-way range scan) --------
    const int g    = tid & (NGROUPS - 1);
    const int part = tid >> 8;                  // 0..3
    const int gs   = g >> 5;                    // sample of this group
    {
        const int lo = s_bound[gs], hi = s_bound[gs + 1];
        const int chunk = ((hi - lo) + NPART - 1) >> 2;
        const int a0 = lo + part * chunk;
        int a1 = a0 + chunk; if (a1 > hi) a1 = hi;
        float cnt = 0.0f, s1 = 0.0f;
        for (int i = a0; i < a1; ++i)
            if (s_key[i] == g) { cnt += s_cnt[i]; s1 += s_val[i]; }
        s_pcnt[part][g] = cnt;
        s_ps[part][g]   = s1;
    }
    __syncthreads();
    if (tid < NGROUPS) {
        // fixed left-assoc combine order -> bit-identical across replays
        const float cnt = s_pcnt[0][tid] + s_pcnt[1][tid] + s_pcnt[2][tid] + s_pcnt[3][tid];
        const float s1  = s_ps[0][tid] + s_ps[1][tid] + s_ps[2][tid] + s_ps[3][tid];
        g_cnt[tid] = cnt;
        g_loc[tid] = (cnt == 0.0f) ? 0.0f : s1 / cnt;     // safe_div
    }
    __syncthreads();

    // ---- Phase 2: q_i = sum_d (t - loc_g)^2 * obs (f32) ------------------
    for (int i = tid; i < S_DIM; i += THREADS) {
        const float loc = g_loc[s_key[i]];
        const float* trow = target + (size_t)(boff + i) * D_DIM;
        const unsigned char* mrow = obs + (size_t)(boff + i) * D_DIM;
        float q = 0.0f;
#pragma unroll
        for (int d = 0; d < D_DIM; ++d) {
            const float df = trow[d] - loc;
            q += df * df * (float)mrow[d];
        }
        s_val[i] = q;
    }
    __syncthreads();

    // ---- Phase 3: s2 -> var = safe_div(s2, cnt - 1), scale ---------------
    {
        const int lo = s_bound[gs], hi = s_bound[gs + 1];
        const int chunk = ((hi - lo) + NPART - 1) >> 2;
        const int a0 = lo + part * chunk;
        int a1 = a0 + chunk; if (a1 > hi) a1 = hi;
        float s2 = 0.0f;
        for (int i = a0; i < a1; ++i)
            if (s_key[i] == g) s2 += s_val[i];
        s_ps[part][g] = s2;
    }
    __syncthreads();
    if (tid < NGROUPS) {
        const float s2  = s_ps[0][tid] + s_ps[1][tid] + s_ps[2][tid] + s_ps[3][tid];
        const float den = g_cnt[tid] - 1.0f;               // CORRECTION = 1
        const float var = (den == 0.0f) ? 0.0f : s2 / den; // safe_div
        g_scale[tid] = sqrtf(var + 1e-5f);                 // MIN_SCALE
    }
    __syncthreads();

    // ---- Phase 4: scatter with pad override (sid==0 <=> key<32) ----------
    for (int i = tid; i < S_DIM; i += THREADS) {
        const int key = s_key[i];
        const bool pad = (key < 32);
        out[boff + i]            = pad ? 0.0f : g_loc[key];
        out[BS_TOTAL + boff + i] = pad ? 1.0f : g_scale[key];
    }
}

// ---------------------------------------------------------------------------
extern "C" void kernel_launch(void* const* d_in, const int* in_sizes, int n_in,
                              void* d_out, int out_size, void* d_ws, size_t ws_size,
                              hipStream_t stream) {
    (void)in_sizes; (void)n_in; (void)out_size; (void)d_ws; (void)ws_size;

    const float*         target = (const float*)d_in[0];
    const unsigned char* obsmsk = (const unsigned char*)d_in[1];   // jax bool: 1B/elem
    const int*           sid    = (const int*)d_in[2];
    const int*           vid    = (const int*)d_in[3];
    float*               out    = (float*)d_out;

    // One block per batch; everything LDS-resident, single launch.
    pss_fused_kernel<<<B_DIM, THREADS, 0, stream>>>(target, obsmsk, sid, vid, out);
}
